// PVBMLIF_54374285968038
// MI455X (gfx1250) — matrix operand order repaired and checked by reading, not compiled, not run on hardware
//
#include <hip/hip_runtime.h>

typedef __attribute__((ext_vector_type(16))) _Float16 v16h;
typedef __attribute__((ext_vector_type(8)))  float    v8f;
typedef __attribute__((ext_vector_type(4)))  float    v4f;

namespace {
constexpr int T_STEPS = 5;
constexpr int BS      = 32;
constexpr int CH      = 64;
constexpr int H       = 64;
constexpr int W       = 64;
constexpr int HW      = H * W;          // 4096
constexpr int NTOT    = T_STEPS * BS;   // 160
}

__device__ __forceinline__ int iabs_(int v) { return v < 0 ? -v : v; }

// ---------------------------------------------------------------------------
// Kernel 1: DropBlock mask via WMMA.
// m = (noise < gamma) in {0,1}.  7x7 SAME max-pool of a 0/1 field == (7x7 box
// sum > 0).  Box sum is separable: S2 = Band * M * Band, Band[i][j] = |i-j|<=3.
// 0/1 inputs and sums <= 49 are exact in f16 -> v_wmma_f32_16x16x32_f16 is
// bit-exact here.  One block (4 wave32) per sample; wave w owns output row
// block w (16 rows), loops over 4 column blocks; K=64 -> 2 WMMA per tile.
// ---------------------------------------------------------------------------
__global__ __launch_bounds__(128) void dropblock_mask_wmma(
    const float* __restrict__ noise, float* __restrict__ bm)
{
  __shared__ _Float16 m_lds[HW];   // bernoulli mask, f16
  __shared__ _Float16 s1_lds[HW];  // horizontal box sum, f16 (integers <= 7)

  const int n   = blockIdx.x;
  const int tid = threadIdx.x;
  constexpr float gamma = 0.1f / 49.0f;

  const float* src = noise + (size_t)n * HW;
  for (int i = tid; i < HW; i += 128)
    m_lds[i] = (src[i] < gamma) ? (_Float16)1.0f : (_Float16)0.0f;
  __syncthreads();

  const int lane   = tid & 31;
  const int r      = lane & 15;   // row / N index within tile
  const int hi     = lane >> 4;   // half-wave select
  const int i_tile = tid >> 5;    // wave id 0..3 -> output row block

  // ---- Phase 2: S1 = M x Band (horizontal 7-wide box sum) ----
  {
    const int arow = i_tile * 16 + r;
    // A fragment (16x32 f16 layout): lane hi=0 holds K 0..7,16..23; hi=1 holds
    // K 8..15,24..31 of row arow.
    v16h a0, a1;
#pragma unroll
    for (int i = 0; i < 16; ++i) {
      const int k = i + hi * 8 + ((i >= 8) ? 8 : 0);
      a0[i] = m_lds[arow * W + k];
      a1[i] = m_lds[arow * W + 32 + k];
    }
#pragma unroll
    for (int j_tile = 0; j_tile < 4; ++j_tile) {
      const int ncol = j_tile * 16 + r;
      // B fragment (32x16): element i -> K = i + 8*hi + (i>=8 ? 8 : 0), N = ncol. Band matrix
      // synthesized in registers (edge clipping == SAME zero padding).
      v16h b0, b1;
#pragma unroll
      for (int i = 0; i < 16; ++i) {
        const int k = i + hi * 8 + ((i >= 8) ? 8 : 0);
        b0[i] = (iabs_(k - ncol)      <= 3) ? (_Float16)1.0f : (_Float16)0.0f;
        b1[i] = (iabs_(k + 32 - ncol) <= 3) ? (_Float16)1.0f : (_Float16)0.0f;
      }
      v8f c = {};
      c = __builtin_amdgcn_wmma_f32_16x16x32_f16(false, a0, false, b0,
                                                 (short)0, c, false, false);
      c = __builtin_amdgcn_wmma_f32_16x16x32_f16(false, a1, false, b1,
                                                 (short)0, c, false, false);
      // C layout: VGPR j -> M = j + 8*hi, N = lane&15
#pragma unroll
      for (int j = 0; j < 8; ++j) {
        const int row = i_tile * 16 + j + hi * 8;
        s1_lds[row * W + j_tile * 16 + r] = (_Float16)c[j];
      }
    }
  }
  __syncthreads();

  // ---- Phase 3: S2 = Band x S1 (vertical pass), then mask = (S2==0) ----
  {
    const int arow = i_tile * 16 + r;
    v16h a0, a1;
#pragma unroll
    for (int i = 0; i < 16; ++i) {
      const int k = i + hi * 8 + ((i >= 8) ? 8 : 0);
      a0[i] = (iabs_(arow - k)        <= 3) ? (_Float16)1.0f : (_Float16)0.0f;
      a1[i] = (iabs_(arow - (k + 32)) <= 3) ? (_Float16)1.0f : (_Float16)0.0f;
    }
#pragma unroll
    for (int j_tile = 0; j_tile < 4; ++j_tile) {
      const int ncol = j_tile * 16 + r;
      v16h b0, b1;
#pragma unroll
      for (int i = 0; i < 16; ++i) {
        const int k = i + hi * 8 + ((i >= 8) ? 8 : 0);
        b0[i] = s1_lds[k * W + ncol];
        b1[i] = s1_lds[(k + 32) * W + ncol];
      }
      v8f c = {};
      c = __builtin_amdgcn_wmma_f32_16x16x32_f16(false, a0, false, b0,
                                                 (short)0, c, false, false);
      c = __builtin_amdgcn_wmma_f32_16x16x32_f16(false, a1, false, b1,
                                                 (short)0, c, false, false);
#pragma unroll
      for (int j = 0; j < 8; ++j) {
        const int row = i_tile * 16 + j + hi * 8;
        const int col = j_tile * 16 + r;
        // block_mask = 1 - pooled = 1 - (boxsum > 0)
        bm[(size_t)n * HW + row * W + col] = (c[j] > 0.5f) ? 0.0f : 1.0f;
      }
    }
  }
}

// ---------------------------------------------------------------------------
// Kernel 2: LIF scan. Bandwidth-bound: 167.8 MB in + 167.8 MB out -> ~14.4 us
// roofline at 23.3 TB/s.  u lives in registers across all 5 timesteps, so x
// and o touch HBM exactly once each.  x/o use non-temporal B128 accesses (no
// reuse); block_mask (2.6 MB, reused by 64 channels) stays regular-temporal in
// L2.  b,c come from blockIdx -> vth[t*32+b] is a block-uniform scalar load.
// ---------------------------------------------------------------------------
__global__ __launch_bounds__(256) void lif_scan(
    const float* __restrict__ x, const float* __restrict__ vth,
    const float* __restrict__ bm, float* __restrict__ out)
{
  const int b   = blockIdx.z;
  const int c   = blockIdx.y;
  const int pix = (blockIdx.x * 256 + threadIdx.x) * 4;  // float4 along W
  const size_t chw = (size_t)CH * HW;

  v4f u = {0.0f, 0.0f, 0.0f, 0.0f};
#pragma unroll
  for (int t = 0; t < T_STEPS; ++t) {
    const int nidx = t * BS + b;
    const float vt = vth[nidx];                         // block-uniform
    const size_t xoff = (size_t)nidx * chw + (size_t)c * HW + (size_t)pix;
    const v4f xv  = __builtin_nontemporal_load((const v4f*)(x + xoff));
    const v4f bmv = *(const v4f*)(bm + (size_t)nidx * HW + pix);
    v4f ov;
#define LIF_COMP(f)                              \
    {                                            \
      const float uu = 0.5f * u.f + xv.f;        \
      const float s  = (uu >= vt) ? 1.0f : 0.0f; \
      ov.f = s * bmv.f;                          \
      u.f  = uu - s * vt;                        \
    }
    LIF_COMP(x) LIF_COMP(y) LIF_COMP(z) LIF_COMP(w)
#undef LIF_COMP
    __builtin_nontemporal_store(ov, (v4f*)(out + xoff));
  }
}

extern "C" void kernel_launch(void* const* d_in, const int* in_sizes, int n_in,
                              void* d_out, int out_size, void* d_ws, size_t ws_size,
                              hipStream_t stream) {
  (void)in_sizes; (void)n_in; (void)out_size; (void)ws_size;
  const float* x     = (const float*)d_in[0];  // [160,64,64,64]
  const float* vth   = (const float*)d_in[1];  // [160]
  const float* noise = (const float*)d_in[2];  // [160,64,64]
  float* out = (float*)d_out;                  // [160,64,64,64]
  float* bm  = (float*)d_ws;                   // scratch: 160*4096 floats = 2.62 MB

  // 1) block mask via WMMA (one block per sample, 4 wave32 each)
  dropblock_mask_wmma<<<dim3(NTOT), dim3(128), 0, stream>>>(noise, bm);
  // 2) streaming LIF scan: grid (HW/1024, C, bs) x 256 threads
  lif_scan<<<dim3(HW / (4 * 256), CH, BS), dim3(256), 0, stream>>>(x, vth, bm, out);
}
